// MultiHeadAttention_8297876816249
// MI455X (gfx1250) — matrix operand order repaired and checked by reading, not compiled, not run on hardware
//
#include <hip/hip_runtime.h>
#include <hip/hip_bf16.h>

typedef __attribute__((ext_vector_type(16))) __bf16 v16bf;
typedef __attribute__((ext_vector_type(8)))  __bf16 v8bf;
typedef __attribute__((ext_vector_type(8)))  float  v8f;
typedef __attribute__((ext_vector_type(4)))  float  v4f;
typedef unsigned int u32x4 __attribute__((ext_vector_type(4)));
typedef int          i32x8 __attribute__((ext_vector_type(8)));
typedef int          i32x4 __attribute__((ext_vector_type(4)));

#if __has_builtin(__builtin_amdgcn_tensor_load_to_lds)
#define HAS_TDM 1
#else
#define HAS_TDM 0
#endif

__device__ __forceinline__ v16bf cat8(v8bf a, v8bf b) {
    return __builtin_shufflevector(a, b, 0,1,2,3,4,5,6,7,8,9,10,11,12,13,14,15);
}

// 16-byte global -> LDS async copy (ASYNCcnt-tracked, bypasses VGPRs).
__device__ __forceinline__ void async_cp16(void* lds_dst, const void* gsrc) {
    uint32_t l = (uint32_t)(uintptr_t)lds_dst;    // ISA 10.2: LDS addr = flat[31:0]
    uint64_t g = (uint64_t)(uintptr_t)gsrc;
    asm volatile("global_load_async_to_lds_b128 %0, %1, off" :: "v"(l), "v"(g) : "memory");
}
#define WAIT_ASYNC(n) asm volatile("s_wait_asynccnt " #n ::: "memory")

#if HAS_TDM
// TDM: DMA a 2-D bf16 tile (rows x row_elems, stride src_stride elems) from
// global into LDS with per-row padding (ISA ch.8 D# layout). Wave-level op;
// call from ONE wave, completion via s_wait_tensorcnt + workgroup barrier.
//   pad_int_c: pad every 2^(c+1) DWORDs; pad_amt_c: pad (c+1) DWORDs.
__device__ __forceinline__ void tdm_load_2d(const void* gsrc, void* lds,
                                            uint32_t row_elems, uint32_t rows,
                                            uint32_t src_stride,
                                            uint32_t pad_int_c, uint32_t pad_amt_c)
{
    uint64_t ga = (uint64_t)(uintptr_t)gsrc;
    u32x4 g0;
    g0[0] = 1u;                                        // count=1 (valid user D#)
    g0[1] = (uint32_t)(uintptr_t)lds;                  // lds_addr
    g0[2] = (uint32_t)ga;                              // global_addr[31:0]
    g0[3] = (uint32_t)(ga >> 32) | (2u << 30);         // global_addr[56:32] | type=2
    i32x8 g1;
    g1[0] = (int)((1u << 16)                           // data_size = 2 bytes
                | (1u << 20)                           // pad_enable
                | (pad_int_c << 22) | (pad_amt_c << 25));
    g1[1] = (int)((src_stride & 0xFFFFu) << 16);       // tensor_dim0[15:0]
    g1[2] = (int)((src_stride >> 16) | ((rows & 0xFFFFu) << 16));   // dim0 hi | tensor_dim1 lo
    g1[3] = (int)(((rows >> 16) & 0xFFFFu) | (row_elems << 16));    // dim1 hi | tile_dim0
    g1[4] = (int)rows;                                 // tile_dim1 (tile_dim2 = 0)
    g1[5] = (int)src_stride;                           // tensor_dim0_stride[31:0]
    g1[6] = 0; g1[7] = 0;
    i32x4 z4 = {0, 0, 0, 0};
    i32x8 z8 = {0, 0, 0, 0, 0, 0, 0, 0};
    // 6-arg form (clang-23 / therock-10.0 headers)
    __builtin_amdgcn_tensor_load_to_lds(g0, g1, z4, z4, z8, 0);
}
#define WAIT_TENSOR(n) __builtin_amdgcn_s_wait_tensorcnt((short)(n))
#endif

// ---------------------------------------------------------------------------
// fp32 -> bf16 elementwise convert (8 elems/thread)
// ---------------------------------------------------------------------------
__global__ __launch_bounds__(256)
void cvt_f32_bf16(const float* __restrict__ src, __bf16* __restrict__ dst, int n)
{
    int i = (blockIdx.x * 256 + threadIdx.x) * 8;
    if (i + 8 <= n) {
        v4f a = *(const v4f*)(src + i);
        v4f b = *(const v4f*)(src + i + 4);
        v8bf o;
#pragma unroll
        for (int e = 0; e < 4; ++e) { o[e] = (__bf16)a[e]; o[4 + e] = (__bf16)b[e]; }
        *(v8bf*)(dst + i) = o;
    }
}

// ---------------------------------------------------------------------------
// All-bf16 tiled GEMM: Y[M,N] = A[M,K] * W[N,K]^T + bias[N]
// 256 threads (8 waves), tile 128x128, K-step 32, double-buffered LDS filled
// by TDM tensor_load_to_lds (wave 0 issues one descriptor per matrix tile).
//   SPLIT_HEADS: bf16 out in [B=2,H=16,S=2048,64]; else fp32 row-major [M,N].
// ---------------------------------------------------------------------------
template <bool SPLIT_HEADS>
__global__ __launch_bounds__(256)
void gemm_bf16_wmma(const __bf16* __restrict__ A, const __bf16* __restrict__ W,
                    const float* __restrict__ bias, void* __restrict__ Op,
                    int M, int N, int K)
{
    __shared__ __bf16 As[2][128 * 40];   // [row][k], row stride 80B = 64B + 16B pad
    __shared__ __bf16 Bs[2][128 * 40];   // [n][k]

    const int tid  = threadIdx.x;
    const int w    = tid >> 5;
    const int lane = tid & 31;
    const int half = lane >> 4;
    const int lann = lane & 15;
    const int wm   = w & 3;
    const int wn   = w >> 2;
    const int gm0  = blockIdx.y * 128;
    const int gn0  = blockIdx.x * 128;

#if HAS_TDM
    // one descriptor per 128x32 tile: row 32 bf16 = 16 DW (pad_int code 3),
    // pad 4 DW (code 3) -> LDS row stride 80B.
    auto issue = [&](int buf, int k0) {
        if (w == 0) {
            tdm_load_2d(A + (size_t)gm0 * K + k0, As[buf], 32, 128, K, 3, 3);
            tdm_load_2d(W + (size_t)gn0 * K + k0, Bs[buf], 32, 128, K, 3, 3);
        }
    };
#else
    auto issue = [&](int buf, int k0) {
#pragma unroll
        for (int j = 0; j < 2; ++j) {
            int fi = tid + j * 256;
            int row = fi >> 2, c8 = fi & 3;
            async_cp16(&As[buf][row * 40 + c8 * 8],
                       A + (size_t)(gm0 + row) * K + k0 + c8 * 8);
            async_cp16(&Bs[buf][row * 40 + c8 * 8],
                       W + (size_t)(gn0 + row) * K + k0 + c8 * 8);
        }
    };
#endif

    v8f c[2][4] = {};
    issue(0, 0);

    const int nIter = K / 32;
    for (int it = 0; it < nIter; ++it) {
        const int cur = it & 1;
        const bool hasNext = (it + 1 < nIter);
        if (hasNext) issue(cur ^ 1, (it + 1) * 32);
#if HAS_TDM
        if (w == 0) { if (hasNext) WAIT_TENSOR(2); else WAIT_TENSOR(0); }
#else
        if (hasNext) { WAIT_ASYNC(0x4); } else { WAIT_ASYNC(0x0); }
#endif
        __syncthreads();                       // current tile visible to all waves

        v16bf a[2], b[4];
#pragma unroll
        for (int mt = 0; mt < 2; ++mt) {
            int row = wm * 32 + mt * 16 + lann;
            v8bf lo = *(const v8bf*)(&As[cur][row * 40 + half * 8]);
            v8bf hi = *(const v8bf*)(&As[cur][row * 40 + 16 + half * 8]);
            a[mt] = cat8(lo, hi);
        }
#pragma unroll
        for (int nt = 0; nt < 4; ++nt) {
            int col = wn * 64 + nt * 16 + lann;
            v8bf lo = *(const v8bf*)(&Bs[cur][col * 40 + half * 8]);
            v8bf hi = *(const v8bf*)(&Bs[cur][col * 40 + 16 + half * 8]);
            b[nt] = cat8(lo, hi);
        }
#pragma unroll
        for (int mt = 0; mt < 2; ++mt)
#pragma unroll
            for (int nt = 0; nt < 4; ++nt)
                c[mt][nt] = __builtin_amdgcn_wmma_f32_16x16x32_bf16(
                    false, a[mt], false, b[nt], (short)0, c[mt][nt], false, false);
        __syncthreads();                       // done reading cur -> reusable
    }

#pragma unroll
    for (int nt = 0; nt < 4; ++nt) {
        int n = gn0 + wn * 64 + nt * 16 + lann;
        float bv = bias[n];
#pragma unroll
        for (int mt = 0; mt < 2; ++mt) {
#pragma unroll
            for (int vr = 0; vr < 8; ++vr) {
                int m = gm0 + wm * 32 + mt * 16 + vr + 8 * half;
                float val = c[mt][nt][vr] + bv;
                if (SPLIT_HEADS) {
                    __bf16* Ob = (__bf16*)Op;
                    int bb = m >> 11, s = m & 2047;     // S = 2048
                    int h = n >> 6, d = n & 63;         // D_K = 64
                    Ob[(((size_t)(bb * 16 + h)) * 2048 + s) * 64 + d] = (__bf16)val;
                } else {
                    float* Of = (float*)Op;
                    Of[(size_t)m * N + n] = val;
                }
            }
        }
    }
}

// ---------------------------------------------------------------------------
// Flash attention over bf16 Q/K/V in [B,H,S,64]; output bf16 [B,S,1024].
// Grid: (S/128, B*H). 8 waves x 16 query rows. Q arrives via async
// global->LDS (ASYNCcnt); K tiles are double-buffered TDM DMAs (TENSORcnt);
// V goes through VGPRs (transposed store).
// ---------------------------------------------------------------------------
__global__ __launch_bounds__(256)
void flash_attn_wmma(const __bf16* __restrict__ Q, const __bf16* __restrict__ Kb,
                     const __bf16* __restrict__ Vb, const int* __restrict__ mask,
                     __bf16* __restrict__ att)
{
    constexpr int S = 2048, DK = 64;
    __shared__ __bf16 Qs[128 * 72];          // [qrow][d], row stride 144B
    __shared__ __bf16 Ks[2][128 * 72];       // [key][d], double buffered
    __shared__ __bf16 Vs[64 * 136];          // [d][key] (transposed)
    __shared__ __bf16 Ps[8 * 16 * 136];      // per-wave P tile [16][128]

    const int tid = threadIdx.x, w = tid >> 5, lane = tid & 31;
    const int half = lane >> 4, lann = lane & 15;
    const int bh = blockIdx.y;
    const int b = bh >> 4, h = bh & 15;
    const int q0 = blockIdx.x * 128;

    // Q tile: per-thread async copies (1024 x 16B chunks, 4 per thread)
    {
        const __bf16* g = Q + ((size_t)bh * S + q0) * DK;
#pragma unroll
        for (int j = 0; j < 4; ++j) {
            int fi = tid + j * 256;
            int row = fi >> 3, c8 = fi & 7;
            async_cp16(Qs + row * 72 + c8 * 8, g + (size_t)row * DK + c8 * 8);
        }
    }

    // K tile kt -> Ks[buf]: 128x64 tile, row 64 bf16 = 32 DW (pad_int code 4),
    // pad 4 DW (code 3) -> LDS row stride 144B.
    auto issueK = [&](int buf, int kt) {
#if HAS_TDM
        if (w == 0)
            tdm_load_2d(Kb + ((size_t)bh * S + kt * 128) * DK, Ks[buf],
                        64, 128, 64, 4, 3);
#else
        const __bf16* g = Kb + ((size_t)bh * S + kt * 128) * DK;
#pragma unroll
        for (int j = 0; j < 4; ++j) {
            int fi = tid + j * 256;
            int row = fi >> 3, c8 = fi & 7;
            async_cp16(Ks[buf] + row * 72 + c8 * 8, g + (size_t)row * DK + c8 * 8);
        }
#endif
    };
    issueK(0, 0);

    float m_run[8], l_run[8];
    v8f acc[4] = {};
#pragma unroll
    for (int r = 0; r < 8; ++r) { m_run[r] = -3.0e38f; l_run[r] = 0.f; }

    __bf16* Pw = &Ps[w * 16 * 136];
    const int mrow0 = q0 + w * 16;
    const int* mp = mask + (size_t)b * S * S;
    constexpr int NT = S / 128;

    for (int kt = 0; kt < NT; ++kt) {
        __syncthreads();                     // prev compute done: Vs, Ks[buf^1] free
        const int cur = kt & 1;
        const bool hasNext = (kt + 1 < NT);
        if (hasNext) issueK(cur ^ 1, kt + 1);

        // V tile through VGPRs, transposed into [d][key]
        const __bf16* Vp = Vb + ((size_t)bh * S + kt * 128) * DK;
#pragma unroll
        for (int j = 0; j < 4; ++j) {
            int fi = tid + j * 256;
            int row = fi >> 3, c8 = fi & 7;
            v8bf vv = *(const v8bf*)(Vp + row * DK + c8 * 8);
#pragma unroll
            for (int e = 0; e < 8; ++e)
                Vs[(c8 * 8 + e) * 136 + row] = vv[e];
        }
#if HAS_TDM
        if (w == 0) { if (hasNext) WAIT_TENSOR(1); else WAIT_TENSOR(0); }
        if (kt == 0) { WAIT_ASYNC(0x0); }            // Q tile landed
#else
        if (hasNext) { WAIT_ASYNC(0x4); } else { WAIT_ASYNC(0x0); }
#endif
        __syncthreads();

        // ---- scores = Q K^T (K-dim 64 -> 2 WMMA k-steps) ----
        v16bf aq[2];
#pragma unroll
        for (int ks = 0; ks < 2; ++ks) {
            int row = w * 16 + lann;
            v8bf lo = *(const v8bf*)(&Qs[row * 72 + ks * 32 + half * 8]);
            v8bf hi = *(const v8bf*)(&Qs[row * 72 + ks * 32 + 16 + half * 8]);
            aq[ks] = cat8(lo, hi);
        }
        v8f sc[8] = {};
#pragma unroll
        for (int nt = 0; nt < 8; ++nt) {
#pragma unroll
            for (int ks = 0; ks < 2; ++ks) {
                int key = nt * 16 + lann;
                v8bf lo = *(const v8bf*)(&Ks[cur][key * 72 + ks * 32 + half * 8]);
                v8bf hi = *(const v8bf*)(&Ks[cur][key * 72 + ks * 32 + 16 + half * 8]);
                v16bf bk = cat8(lo, hi);
                sc[nt] = __builtin_amdgcn_wmma_f32_16x16x32_bf16(
                    false, aq[ks], false, bk, (short)0, sc[nt], false, false);
            }
        }

        // ---- scale + mask ----
#pragma unroll
        for (int nt = 0; nt < 8; ++nt) {
            int sk = kt * 128 + nt * 16 + lann;
#pragma unroll
            for (int r = 0; r < 8; ++r) {
                int sq = mrow0 + r + 8 * half;
                int mv = mp[(size_t)sq * S + sk];
                sc[nt][r] = (mv == 0) ? -1.0e9f : sc[nt][r] * 0.125f;
            }
        }

        // ---- online softmax (rows live across 16-lane groups) ----
#pragma unroll
        for (int r = 0; r < 8; ++r) {
            float mx = m_run[r];
#pragma unroll
            for (int nt = 0; nt < 8; ++nt) mx = fmaxf(mx, sc[nt][r]);
            mx = fmaxf(mx, __shfl_xor(mx, 1, 16));
            mx = fmaxf(mx, __shfl_xor(mx, 2, 16));
            mx = fmaxf(mx, __shfl_xor(mx, 4, 16));
            mx = fmaxf(mx, __shfl_xor(mx, 8, 16));
            float a_ = __expf(m_run[r] - mx);
            float s_ = 0.f;
#pragma unroll
            for (int nt = 0; nt < 8; ++nt) {
                float e = __expf(sc[nt][r] - mx);
                sc[nt][r] = e;
                s_ += e;
            }
            s_ += __shfl_xor(s_, 1, 16);
            s_ += __shfl_xor(s_, 2, 16);
            s_ += __shfl_xor(s_, 4, 16);
            s_ += __shfl_xor(s_, 8, 16);
            l_run[r] = l_run[r] * a_ + s_;
            m_run[r] = mx;
#pragma unroll
            for (int dt = 0; dt < 4; ++dt) acc[dt][r] *= a_;
        }

        // ---- P -> A-fragment layout via wave-private LDS ----
#pragma unroll
        for (int nt = 0; nt < 8; ++nt)
#pragma unroll
            for (int r = 0; r < 8; ++r)
                Pw[(r + 8 * half) * 136 + nt * 16 + lann] = (__bf16)sc[nt][r];
        asm volatile("s_wait_dscnt 0x0" ::: "memory");

        // ---- acc += P * V (K-dim 128 keys -> 4 WMMA k-steps) ----
#pragma unroll
        for (int kk = 0; kk < 4; ++kk) {
            v8bf plo = *(const v8bf*)(&Pw[lann * 136 + kk * 32 + half * 8]);
            v8bf phi = *(const v8bf*)(&Pw[lann * 136 + kk * 32 + 16 + half * 8]);
            v16bf pf = cat8(plo, phi);
#pragma unroll
            for (int dt = 0; dt < 4; ++dt) {
                int d = dt * 16 + lann;
                v8bf vlo = *(const v8bf*)(&Vs[d * 136 + kk * 32 + half * 8]);
                v8bf vhi = *(const v8bf*)(&Vs[d * 136 + kk * 32 + 16 + half * 8]);
                v16bf vf = cat8(vlo, vhi);
                acc[dt] = __builtin_amdgcn_wmma_f32_16x16x32_bf16(
                    false, pf, false, vf, (short)0, acc[dt], false, false);
            }
        }
    }

    // ---- normalize + store bf16 att[B,S,1024] ----
#pragma unroll
    for (int dt = 0; dt < 4; ++dt) {
#pragma unroll
        for (int r = 0; r < 8; ++r) {
            int sq = mrow0 + r + 8 * half;
            int d = dt * 16 + lann;
            float o = acc[dt][r] / l_run[r];
            att[((size_t)b * S + sq) * 1024 + h * 64 + d] = (__bf16)o;
        }
    }
}

// ---------------------------------------------------------------------------
extern "C" void kernel_launch(void* const* d_in, const int* in_sizes, int n_in,
                              void* d_out, int out_size, void* d_ws, size_t ws_size,
                              hipStream_t stream)
{
    (void)in_sizes; (void)n_in; (void)out_size; (void)ws_size;
    const float* q    = (const float*)d_in[0];
    const float* k    = (const float*)d_in[1];
    const float* v    = (const float*)d_in[2];
    const int*   mask = (const int*)d_in[3];
    const float* wQw  = (const float*)d_in[4];
    const float* wQb  = (const float*)d_in[5];
    const float* wKw  = (const float*)d_in[6];
    const float* wKb  = (const float*)d_in[7];
    const float* wVw  = (const float*)d_in[8];
    const float* wVb  = (const float*)d_in[9];
    const float* wOw  = (const float*)d_in[10];
    const float* wOb  = (const float*)d_in[11];
    float* out = (float*)d_out;

    char* ws = (char*)d_ws;
    const size_t MB = (size_t)1 << 20;
    __bf16* qc   = (__bf16*)(ws);             // [B,S,1024] bf16, 8 MiB
    __bf16* kc   = (__bf16*)(ws + 8  * MB);
    __bf16* vc   = (__bf16*)(ws + 16 * MB);
    __bf16* wQc  = (__bf16*)(ws + 24 * MB);   // [1024,1024] bf16, 2 MiB each
    __bf16* wKc  = (__bf16*)(ws + 26 * MB);
    __bf16* wVc  = (__bf16*)(ws + 28 * MB);
    __bf16* wOc  = (__bf16*)(ws + 30 * MB);
    __bf16* qh   = (__bf16*)(ws + 32 * MB);   // head-split [B,H,S,64], 8 MiB each
    __bf16* kh   = (__bf16*)(ws + 40 * MB);
    __bf16* vh   = (__bf16*)(ws + 48 * MB);
    __bf16* attb = (__bf16*)(ws + 56 * MB);

    const int M = 4096, N = 1024, K = 1024;   // M = B*S
    const int nIn = M * K, nW = N * K;

    cvt_f32_bf16<<<nIn / 2048, 256, 0, stream>>>(q,   qc,  nIn);
    cvt_f32_bf16<<<nIn / 2048, 256, 0, stream>>>(k,   kc,  nIn);
    cvt_f32_bf16<<<nIn / 2048, 256, 0, stream>>>(v,   vc,  nIn);
    cvt_f32_bf16<<<nW  / 2048, 256, 0, stream>>>(wQw, wQc, nW);
    cvt_f32_bf16<<<nW  / 2048, 256, 0, stream>>>(wKw, wKc, nW);
    cvt_f32_bf16<<<nW  / 2048, 256, 0, stream>>>(wVw, wVc, nW);
    cvt_f32_bf16<<<nW  / 2048, 256, 0, stream>>>(wOw, wOc, nW);

    dim3 gg(N / 128, M / 128), bb(256);
    gemm_bf16_wmma<true><<<gg, bb, 0, stream>>>(qc, wQc, wQb, qh, M, N, K);
    gemm_bf16_wmma<true><<<gg, bb, 0, stream>>>(kc, wKc, wKb, kh, M, N, K);
    gemm_bf16_wmma<true><<<gg, bb, 0, stream>>>(vc, wVc, wVb, vh, M, N, K);

    flash_attn_wmma<<<dim3(16, 32), 256, 0, stream>>>(qh, kh, vh, mask, attb);

    gemm_bf16_wmma<false><<<gg, bb, 0, stream>>>(attb, wOc, wOb, out, M, N, K);
}
